// MastishkAttention_28089086116077
// MI455X (gfx1250) — matrix operand order repaired and checked by reading, not compiled, not run on hardware
//
#include <hip/hip_runtime.h>
#include <hip/hip_bf16.h>

// ---------------------------------------------------------------------------
// MHA forward for MI455X (gfx1250): bf16 WMMA everywhere, f32 accumulate.
// B=2, S=2048, H=1024, NH=16, HD=64.
// GEMM stages the shared W panel through LDS via gfx1250 async
// global->LDS b128 copies (ASYNCcnt-tracked).
// ---------------------------------------------------------------------------

typedef __attribute__((ext_vector_type(16))) __bf16 bf16x16;
typedef __attribute__((ext_vector_type(8)))  __bf16 bf16x8;
typedef __attribute__((ext_vector_type(8)))  float  f32x8;
typedef __attribute__((ext_vector_type(4)))  int    v4i;

typedef __attribute__((address_space(1))) v4i gas_v4i;   // global int4
typedef __attribute__((address_space(3))) v4i las_v4i;   // LDS int4

union BF16x16 { bf16x8 h[2]; bf16x16 v; };

__device__ __forceinline__ f32x8 wmma_bf16(const BF16x16& a, const BF16x16& b, f32x8 c) {
    // D = A(16x32 bf16) x B(32x16 bf16) + C(16x16 f32)
    return __builtin_amdgcn_wmma_f32_16x16x32_bf16(
        /*neg_a=*/false, a.v, /*neg_b=*/false, b.v,
        /*c_mod=*/(short)0, c, /*reuse_a=*/false, /*reuse_b=*/false);
}

#if defined(__gfx1250__) && __has_builtin(__builtin_amdgcn_global_load_async_to_lds_b128)
#define USE_ASYNC_LDS 1
#else
#define USE_ASYNC_LDS 0
#endif

// ------------------------------- f32 -> bf16 -------------------------------
__global__ void cvt_f32_bf16(const float* __restrict__ in, __bf16* __restrict__ out, int n) {
    int i = blockIdx.x * blockDim.x + threadIdx.x;
    if (i < n) out[i] = (__bf16)in[i];
}

// -------------------- GEMM: Y[M,N] = A[M,K] @ W[N,K]^T ---------------------
// Per-wave 16(M) x 64(N) tile; 8 waves/block -> 128x64 block tile.
// The 64x32 W panel (shared by all 8 waves) is staged through LDS per k-step.
template <bool BF16OUT>
__global__ __launch_bounds__(256) void gemm_xwT(
    const __bf16* __restrict__ A, const __bf16* __restrict__ W,
    void* __restrict__ outp, int M, int N, int K) {
    const int tid  = threadIdx.x;
    const int lane = tid & 31;
    const int wave = tid >> 5;
    const int m0 = blockIdx.x * 128 + wave * 16;
    const int n0 = blockIdx.y * 64;
    const int rc = lane & 15;       // A-row / B-col within tile
    const int hi = lane >> 4;
    const int kA = hi * 16;         // A-matrix K base (contiguous k = 16h + s)
    const int kB = hi * 16;         // B-matrix K base (contiguous k = 16h + s)

    __shared__ __align__(16) __bf16 Wlds[64 * 32];   // 4 KB W panel

    // cooperative W-panel staging: 256 threads x 16B = 64 rows x 32 cols bf16
    const int lr = tid >> 2;            // panel row 0..63
    const int lc = (tid & 3) * 8;       // panel col offset (elements)
    const __bf16* gsrc = W + (size_t)(n0 + lr) * K + lc;
    __bf16* ldst = &Wlds[lr * 32 + lc];

    f32x8 acc0 = {}, acc1 = {}, acc2 = {}, acc3 = {};

    const __bf16* ap = A + (size_t)(m0 + rc) * K;
    const __bf16* b0 = &Wlds[(0 * 16 + rc) * 32 + kB];
    const __bf16* b1 = &Wlds[(1 * 16 + rc) * 32 + kB];
    const __bf16* b2 = &Wlds[(2 * 16 + rc) * 32 + kB];
    const __bf16* b3 = &Wlds[(3 * 16 + rc) * 32 + kB];

    for (int kt = 0; kt < K; kt += 32) {
        __syncthreads();               // WAR: previous iteration's LDS reads done
#if USE_ASYNC_LDS
        __builtin_amdgcn_global_load_async_to_lds_b128(
            (gas_v4i*)(gsrc + kt), (las_v4i*)ldst, 0, 0);
#else
        *(bf16x8*)ldst = *(const bf16x8*)(gsrc + kt);
#endif
        __builtin_prefetch(ap + kt + 128, 0, 0);   // global_prefetch_b8

        // A fragment straight from global (L2-resident), contiguous k = 16h + s
        BF16x16 a;
        a.h[0] = *(const bf16x8*)(ap + kt + kA);
        a.h[1] = *(const bf16x8*)(ap + kt + kA + 8);

#if USE_ASYNC_LDS
#if __has_builtin(__builtin_amdgcn_s_wait_asynccnt)
        __builtin_amdgcn_s_wait_asynccnt(0);
#else
        asm volatile("s_wait_asynccnt 0" ::: "memory");
#endif
#endif
        __syncthreads();               // W panel visible to all waves

        BF16x16 b;
        b.h[0] = *(const bf16x8*)(b0);
        b.h[1] = *(const bf16x8*)(b0 + 8);
        acc0 = wmma_bf16(a, b, acc0);
        b.h[0] = *(const bf16x8*)(b1);
        b.h[1] = *(const bf16x8*)(b1 + 8);
        acc1 = wmma_bf16(a, b, acc1);
        b.h[0] = *(const bf16x8*)(b2);
        b.h[1] = *(const bf16x8*)(b2 + 8);
        acc2 = wmma_bf16(a, b, acc2);
        b.h[0] = *(const bf16x8*)(b3);
        b.h[1] = *(const bf16x8*)(b3 + 8);
        acc3 = wmma_bf16(a, b, acc3);
    }

    const int gN = n0 + rc;
    if (BF16OUT) {
        __bf16* out = (__bf16*)outp;
#pragma unroll
        for (int r = 0; r < 8; ++r) {
            size_t base = (size_t)(m0 + r + hi * 8) * N + gN;
            out[base +  0] = (__bf16)acc0[r];
            out[base + 16] = (__bf16)acc1[r];
            out[base + 32] = (__bf16)acc2[r];
            out[base + 48] = (__bf16)acc3[r];
        }
    } else {
        float* out = (float*)outp;
#pragma unroll
        for (int r = 0; r < 8; ++r) {
            size_t base = (size_t)(m0 + r + hi * 8) * N + gN;
            out[base +  0] = acc0[r];
            out[base + 16] = acc1[r];
            out[base + 32] = acc2[r];
            out[base + 48] = acc3[r];
        }
    }
}

// ---------------- V transpose: Vb[b*S+s, h*64+d] -> Vt[b,h,d,s] ------------
__global__ void transpose_v(const __bf16* __restrict__ Vb, __bf16* __restrict__ Vt) {
    size_t i = (size_t)blockIdx.x * blockDim.x + threadIdx.x;   // over 4096*1024
    int n = (int)(i & 1023);
    int m = (int)(i >> 10);
    int s = m & 2047;
    int b = m >> 11;
    int h = n >> 6;
    int d = n & 63;
    Vt[(((size_t)(b * 16 + h) * 64 + d) << 11) + s] = Vb[i];
}

// -------------------------- flash attention core ---------------------------
// One wave per (b, h, 16-query tile).  S=2048, H=1024, NH=16, HD=64.
__global__ __launch_bounds__(32) void attn_fwd(
    const __bf16* __restrict__ Qb, const __bf16* __restrict__ Kb,
    const __bf16* __restrict__ Vt, __bf16* __restrict__ Ob) {
    constexpr int S = 2048, Hd = 1024, NH = 16, HD = 64;
    const int lane = threadIdx.x;
    const int qt = blockIdx.x & 127;
    const int h  = (blockIdx.x >> 7) & 15;
    const int b  = blockIdx.x >> 11;
    const int q0 = qt * 16;
    const int rc = lane & 15;
    const int hi = lane >> 4;
    const int kA = hi * 16;
    const int kB = hi * 16;
    const int hiM = hi * 8;

    __shared__ __align__(16) __bf16 Plds[16 * 32];

    // Q tile, A-layout, d = 0..31 and 32..63
    const __bf16* qp = Qb + ((size_t)(b * S + q0 + rc) * Hd + h * HD);
    BF16x16 qa0, qa1;
    qa0.h[0] = *(const bf16x8*)(qp + kA);
    qa0.h[1] = *(const bf16x8*)(qp + kA + 8);
    qa1.h[0] = *(const bf16x8*)(qp + 32 + kA);
    qa1.h[1] = *(const bf16x8*)(qp + 32 + kA + 8);

    float mrow[8], lrow[8];
#pragma unroll
    for (int r = 0; r < 8; ++r) { mrow[r] = -1e30f; lrow[r] = 0.0f; }
    f32x8 o0 = {}, o1 = {}, o2 = {}, o3 = {};

    for (int j0 = 0; j0 <= q0 + 15; j0 += 32) {
        // ---- scores: S0 = Q . K^T (keys j0..j0+15), S1 = keys j0+16..j0+31
        f32x8 s0 = {}, s1 = {};
        {
            const __bf16* kp = Kb + ((size_t)(b * S + j0 + rc) * Hd + h * HD) + kB;
            BF16x16 kb;
            kb.h[0] = *(const bf16x8*)(kp);      kb.h[1] = *(const bf16x8*)(kp + 8);
            s0 = wmma_bf16(qa0, kb, s0);
            kb.h[0] = *(const bf16x8*)(kp + 32); kb.h[1] = *(const bf16x8*)(kp + 40);
            s0 = wmma_bf16(qa1, kb, s0);
        }
        {
            const __bf16* kp = Kb + ((size_t)(b * S + j0 + 16 + rc) * Hd + h * HD) + kB;
            BF16x16 kb;
            kb.h[0] = *(const bf16x8*)(kp);      kb.h[1] = *(const bf16x8*)(kp + 8);
            s1 = wmma_bf16(qa0, kb, s1);
            kb.h[0] = *(const bf16x8*)(kp + 32); kb.h[1] = *(const bf16x8*)(kp + 40);
            s1 = wmma_bf16(qa1, kb, s1);
        }

        __syncthreads();
        // ---- online softmax over the 32-key block (row lives across a 16-lane half)
#pragma unroll
        for (int r = 0; r < 8; ++r) {
            const int q = q0 + r + hiM;
            float v0 = s0[r] * 0.125f;                // 1/sqrt(64)
            if (j0 + rc > q)      v0 = -1e30f;        // causal mask
            float v1 = s1[r] * 0.125f;
            if (j0 + 16 + rc > q) v1 = -1e30f;
            float t = fmaxf(v0, v1);
            t = fmaxf(t, __shfl_xor(t, 1));
            t = fmaxf(t, __shfl_xor(t, 2));
            t = fmaxf(t, __shfl_xor(t, 4));
            t = fmaxf(t, __shfl_xor(t, 8));
            float mnew  = fmaxf(mrow[r], t);
            float alpha = __expf(mrow[r] - mnew);
            float p0 = __expf(v0 - mnew);
            float p1 = __expf(v1 - mnew);
            float rs = p0 + p1;
            rs += __shfl_xor(rs, 1);
            rs += __shfl_xor(rs, 2);
            rs += __shfl_xor(rs, 4);
            rs += __shfl_xor(rs, 8);
            lrow[r] = lrow[r] * alpha + rs;
            mrow[r] = mnew;
            o0[r] *= alpha; o1[r] *= alpha; o2[r] *= alpha; o3[r] *= alpha;
            // stage P (C-layout -> row-major LDS) for re-read in A-layout
            Plds[(r + hiM) * 32 + rc]      = (__bf16)p0;
            Plds[(r + hiM) * 32 + 16 + rc] = (__bf16)p1;
        }
        __syncthreads();

        // ---- O += P(16x32) . V(32x64)
        BF16x16 pa;
        pa.h[0] = *(const bf16x8*)(&Plds[rc * 32 + kA]);
        pa.h[1] = *(const bf16x8*)(&Plds[rc * 32 + kA + 8]);
        const __bf16* vp = Vt + ((size_t)((b * NH + h) * HD + rc) * S + j0 + kB);
        BF16x16 vb;
        vb.h[0] = *(const bf16x8*)(vp);              vb.h[1] = *(const bf16x8*)(vp + 8);
        o0 = wmma_bf16(pa, vb, o0);
        vb.h[0] = *(const bf16x8*)(vp + 16 * S);     vb.h[1] = *(const bf16x8*)(vp + 16 * S + 8);
        o1 = wmma_bf16(pa, vb, o1);
        vb.h[0] = *(const bf16x8*)(vp + 32 * S);     vb.h[1] = *(const bf16x8*)(vp + 32 * S + 8);
        o2 = wmma_bf16(pa, vb, o2);
        vb.h[0] = *(const bf16x8*)(vp + 48 * S);     vb.h[1] = *(const bf16x8*)(vp + 48 * S + 8);
        o3 = wmma_bf16(pa, vb, o3);
    }

    // ---- normalize + emit O (bf16, [B*S, H] layout for the out-projection)
#pragma unroll
    for (int r = 0; r < 8; ++r) {
        float inv = 1.0f / lrow[r];
        size_t base = (size_t)(b * S + q0 + r + hiM) * Hd + h * HD + rc;
        Ob[base +  0] = (__bf16)(o0[r] * inv);
        Ob[base + 16] = (__bf16)(o1[r] * inv);
        Ob[base + 32] = (__bf16)(o2[r] * inv);
        Ob[base + 48] = (__bf16)(o3[r] * inv);
    }
}

// ---------------------------------------------------------------------------
extern "C" void kernel_launch(void* const* d_in, const int* in_sizes, int n_in,
                              void* d_out, int out_size, void* d_ws, size_t ws_size,
                              hipStream_t stream) {
    (void)in_sizes; (void)n_in; (void)out_size; (void)ws_size;
    const float* X  = (const float*)d_in[0];
    // d_in[1] = attention_mask (deterministic causal mask, applied analytically)
    const float* Wq = (const float*)d_in[2];
    const float* Wk = (const float*)d_in[3];
    const float* Wv = (const float*)d_in[4];
    const float* Wo = (const float*)d_in[5];

    constexpr int M = 4096;   // B*S
    constexpr int N = 1024;   // H
    constexpr int K = 1024;   // H
    const size_t MB = 1024 * 1024;
    char* ws = (char*)d_ws;
    __bf16* Xb  = (__bf16*)(ws +  0 * MB);   // 8 MB
    __bf16* Wqb = (__bf16*)(ws +  8 * MB);   // 2 MB
    __bf16* Wkb = (__bf16*)(ws + 10 * MB);
    __bf16* Wvb = (__bf16*)(ws + 12 * MB);
    __bf16* Wob = (__bf16*)(ws + 14 * MB);
    __bf16* Qb  = (__bf16*)(ws + 16 * MB);   // 8 MB each
    __bf16* Kb  = (__bf16*)(ws + 24 * MB);
    __bf16* Vb  = (__bf16*)(ws + 32 * MB);
    __bf16* Vt  = (__bf16*)(ws + 40 * MB);
    __bf16* Ob  = (__bf16*)(ws + 48 * MB);

    // 1) convert inputs/weights to bf16
    cvt_f32_bf16<<<(M * K) / 256, 256, 0, stream>>>(X,  Xb,  M * K);
    cvt_f32_bf16<<<(N * K) / 256, 256, 0, stream>>>(Wq, Wqb, N * K);
    cvt_f32_bf16<<<(N * K) / 256, 256, 0, stream>>>(Wk, Wkb, N * K);
    cvt_f32_bf16<<<(N * K) / 256, 256, 0, stream>>>(Wv, Wvb, N * K);
    cvt_f32_bf16<<<(N * K) / 256, 256, 0, stream>>>(Wo, Wob, N * K);

    // 2) Q/K/V projections (bf16 outputs)
    dim3 gg(M / 128, N / 64);
    gemm_xwT<true><<<gg, 256, 0, stream>>>(Xb, Wqb, (void*)Qb, M, N, K);
    gemm_xwT<true><<<gg, 256, 0, stream>>>(Xb, Wkb, (void*)Kb, M, N, K);
    gemm_xwT<true><<<gg, 256, 0, stream>>>(Xb, Wvb, (void*)Vb, M, N, K);

    // 3) V -> Vt[b,h,d,s] so P.V B-matrix loads are contiguous along keys
    transpose_v<<<(M * N) / 256, 256, 0, stream>>>(Vb, Vt);

    // 4) causal flash attention: one wave per (b, h, 16-query tile)
    attn_fwd<<<2 * 16 * 128, 32, 0, stream>>>(Qb, Kb, Vt, Ob);

    // 5) output projection (f32 result straight to d_out)
    gemm_xwT<false><<<gg, 256, 0, stream>>>(Ob, Wob, d_out, M, N, K);
}
